// GPConv7x7PipelineB_6691559047412
// MI455X (gfx1250) — matrix operand order repaired and checked by reading, not compiled, not run on hardware
//
#include <hip/hip_runtime.h>
#include <stdint.h>

typedef __attribute__((ext_vector_type(16))) _Float16 v16h;
typedef __attribute__((ext_vector_type(8)))  float    v8f;

#define BATCH 8
#define HH 512
#define WW 512
#define HWSZ (HH*WW)
#define PI_F 3.14159265358979323846f
#define SSTR (512.0f/511.0f)

// ---------------- Kernel 1: Fm = |x| over channels, G = x0 + x1 ----------------
__global__ __launch_bounds__(256) void fm_g_kernel(const float* __restrict__ x,
                                                   float* __restrict__ Fm,
                                                   float* __restrict__ G) {
  int idx = blockIdx.x * 256 + threadIdx.x;
  if (idx >= BATCH * HWSZ) return;
  int b  = idx >> 18;           // HWSZ == 2^18
  int hw = idx & (HWSZ - 1);
  const float* xb = x + (size_t)b * 2 * HWSZ + hw;
  float x0 = xb[0], x1 = xb[HWSZ];
  Fm[idx] = sqrtf(x0 * x0 + x1 * x1);
  G[idx]  = x0 + x1;
}

// CDNA5 async copy: global -> LDS (per-lane addresses), tracked by ASYNCcnt.
__device__ __forceinline__ void async_lds_load_b32(unsigned lds_off, const float* gptr) {
  asm volatile("global_load_async_to_lds_b32 %0, %1, off"
               :: "v"(lds_off), "v"((unsigned long long)gptr)
               : "memory");
}

// ---------------- Kernel 2: fused line-energy + WMMA conv + routing ----------------
__global__ __launch_bounds__(256) void fused_kernel(
    const float* __restrict__ Fm, const float* __restrict__ G,
    const float* __restrict__ Wb,       // (4,1,7,7) floats
    const float* __restrict__ centers,  // 4 floats
    const float* __restrict__ gain_p, const float* __restrict__ lrt_p,
    const float* __restrict__ rho_p,  float* __restrict__ out)
{
  __shared__ float    Fms[24 * 24];    // Fm tile, halo 4
  __shared__ float    Gs[22 * 22];     // G tile, halo 3 (zero-padded at image edge)
  __shared__ _Float16 Wh[16 * 64];     // conv weights, M padded 4->16, K padded 49->64 (zeros)
  __shared__ float    ybuf[4 * 16 * 16];
  __shared__ float    cq[24], sq[24];
  __shared__ float    cxt[192], cyt[192];  // per (q,n) sample-center offsets
  __shared__ int      offt[64];            // tap -> LDS offset within G tile (0 for pad taps)

  const int tx  = threadIdx.x, ty = threadIdx.y;
  const int tid = ty * 16 + tx;
  const int b   = blockIdx.z;
  const int tj0 = blockIdx.x * 16;
  const int ti0 = blockIdx.y * 16;

  // ---- tables ----
  if (tid < 192) {
    int q = tid >> 3, n = tid & 7;
    float th = (2.0f * PI_F / 24.0f) * (float)q;
    float s, c; __sincosf(th, &s, &c);
    float t = 3.0f * (float)(n + 1) / 8.0f;
    cxt[tid] = t * c * SSTR + 0.5f;
    cyt[tid] = t * s * SSTR + 0.5f;
    if (n == 0) { cq[q] = c; sq[q] = s; }
  }
  // pad taps point at offset 0: the corresponding A weights are zero, so the
  // (finite) garbage contributes nothing -> unconditional LDS reads, no branches.
  if (tid < 64) offt[tid] = (tid < 49) ? (tid / 7) * 22 + (tid % 7) : 0;
  for (int i = tid; i < 16 * 64; i += 256) {
    int m = i >> 6, kg = i & 63;
    Wh[i] = (m < 4 && kg < 49) ? (_Float16)Wb[m * 49 + kg] : (_Float16)0.0f;
  }

  const float* FmB = Fm + (size_t)b * HWSZ;
  const float* GB  = G  + (size_t)b * HWSZ;

  // ---- async stage tiles into LDS ----
  for (int s = tid; s < 576; s += 256) {          // Fm tile: clamp (clipped sample
    int ry = s / 24, rx = s - ry * 24;            // indices never read OOB slots)
    int gy = min(max(ti0 - 4 + ry, 0), HH - 1);
    int gx = min(max(tj0 - 4 + rx, 0), WW - 1);
    async_lds_load_b32((unsigned)(uintptr_t)&Fms[s], FmB + gy * WW + gx);
  }
  for (int s = tid; s < 484; s += 256) {          // G tile: zero pad (conv pads with 0)
    int ry = s / 22, rx = s - ry * 22;
    int gy = ti0 - 3 + ry, gx = tj0 - 3 + rx;
    if (gy >= 0 && gy < HH && gx >= 0 && gx < WW)
      async_lds_load_b32((unsigned)(uintptr_t)&Gs[s], GB + gy * WW + gx);
    else
      Gs[s] = 0.0f;
  }
  asm volatile("s_wait_asynccnt 0x0" ::: "memory");
  __syncthreads();

  // ---- WMMA conv: D(16 orient x 16 px) = A(16x32 wts) * B(32x16 taps), K=49 pad 64 ----
  const int  lane    = (ty & 1) * 16 + tx;   // linear%32
  const bool hihalf  = lane >= 16;
  const int  nn      = lane & 15;            // pixel column (N) / weight row (M)
  const int  ty0     = ty & ~1;              // first pixel row owned by this wave

  v16h A0, A1;
#pragma unroll
  for (int e = 0; e < 16; ++e) {
    // 16-bit A 16x32 layout: K = e + (hi?8:0) + (e>=8?8:0)
    int k = e + (hihalf ? 8 : 0) + (e >= 8 ? 8 : 0);
    A0[e] = Wh[nn * 64 + k];
    A1[e] = Wh[nn * 64 + 32 + k];
  }
#pragma unroll
  for (int r = 0; r < 2; ++r) {
    int rr = ty0 + r;
    int gbase = rr * 22 + nn;
    v16h B0, B1;
#pragma unroll
    for (int e = 0; e < 16; ++e) {
      // 16-bit B 32x16 layout: K = e + (hi?8:0) + (e>=8?8:0)
      int k = e + (hihalf ? 8 : 0) + (e >= 8 ? 8 : 0);
      B0[e] = (_Float16)Gs[gbase + offt[k]];
      B1[e] = (_Float16)Gs[gbase + offt[32 + k]];
    }
    v8f acc = {};
    acc = __builtin_amdgcn_wmma_f32_16x16x32_f16(false, A0, false, B0, (short)0, acc, false, false);
    acc = __builtin_amdgcn_wmma_f32_16x16x32_f16(false, A1, false, B1, (short)0, acc, false, false);
    if (!hihalf) {                    // lanes 0-15 hold M=0..7 in v[0..7]; orientations 0..3
#pragma unroll
      for (int o = 0; o < 4; ++o) ybuf[o * 256 + rr * 16 + nn] = acc[o];
    }
  }
  __syncthreads();

  // ---- per-pixel line energy + online softmax over 24 angles ----
  const int gi = ti0 + ty, gj = tj0 + tx;
  const float pis = (float)gi * SSTR;
  const float pjs = (float)gj * SSTR;

  float wnl[8]; float wsum = 0.f;
#pragma unroll
  for (int n = 0; n < 8; ++n) {
    float t = 3.0f * (float)(n + 1) / 8.0f;
    wnl[n] = __expf(-0.5f * t * t);
    wsum += wnl[n];
  }
  const float invws = 1.0f / wsum;

  float mrun = -1e30f, sy = 0.f, cxs = 0.f;
  for (int q = 0; q < 24; ++q) {
    float acc = 0.f;
#pragma unroll
    for (int n = 0; n < 8; ++n) {
      float px = pjs - cxt[q * 8 + n];
      float py = pis - cyt[q * 8 + n];
      float fx = floorf(px), fy = floorf(py);
      int ix0 = (int)fx, iy0 = (int)fy;
      float ax = px - fx, ay = py - fy;
      float wx0 = (ix0 >= 0  && ix0 < WW)     ? 1.f - ax : 0.f;
      float wx1 = (ix0 >= -1 && ix0 < WW - 1) ? ax       : 0.f;
      float wy0 = (iy0 >= 0  && iy0 < HH)     ? 1.f - ay : 0.f;
      float wy1 = (iy0 >= -1 && iy0 < HH - 1) ? ay       : 0.f;
      int lx0 = min(max(ix0,     0), WW - 1) - tj0 + 4;
      int lx1 = min(max(ix0 + 1, 0), WW - 1) - tj0 + 4;
      int ly0 = min(max(iy0,     0), HH - 1) - ti0 + 4;
      int ly1 = min(max(iy0 + 1, 0), HH - 1) - ti0 + 4;
      float F00 = Fms[ly0 * 24 + lx0], F01 = Fms[ly0 * 24 + lx1];
      float F10 = Fms[ly1 * 24 + lx0], F11 = Fms[ly1 * 24 + lx1];
      acc += wnl[n] * (wy0 * (wx0 * F00 + wx1 * F01) + wy1 * (wx0 * F10 + wx1 * F11));
    }
    float Eq = acc * invws;
    float mo = mrun;
    mrun = fmaxf(mrun, Eq);
    float scale = __expf(mo - mrun);
    float eq    = __expf(Eq - mrun);
    sy  = sy  * scale + sq[q] * eq;
    cxs = cxs * scale + cq[q] * eq;
  }
  float hat = atan2f(sy, cxs);
  float th  = hat - PI_F * floorf(hat / PI_F);   // mod pi, in [0, pi)

  // ---- routing ----
  float rt  = __expf(lrt_p[0]); rt = fminf(fmaxf(rt, 1e-3f), 2.0f);
  float rho = 1.0f / (1.0f + __expf(-rho_p[0])) * 0.95f + 0.05f;
  float gamma = 1.0f / (rho + 1e-8f);
  float g = fmaxf(gain_p[0], 1e-3f);

  float logit[4]; float lm = -1e30f;
#pragma unroll
  for (int o = 0; o < 4; ++o) {
    float d = th - centers[o] + 0.5f * PI_F;
    d -= PI_F * floorf(d / PI_F);
    d = fabsf(d - 0.5f * PI_F);
    logit[o] = -d / rt;
    lm = fmaxf(lm, logit[o]);
  }
  float bs = 0.f, base[4];
#pragma unroll
  for (int o = 0; o < 4; ++o) { base[o] = __expf(logit[o] - lm); bs += base[o]; }
  float ss = 0.f, sharp[4];
#pragma unroll
  for (int o = 0; o < 4; ++o) {
    float bo = base[o] / bs + 1e-12f;
    sharp[o] = __expf(gamma * __logf(bo));
    ss += sharp[o];
  }
  float accum = 0.f;
#pragma unroll
  for (int o = 0; o < 4; ++o)
    accum += ybuf[o * 256 + ty * 16 + tx] * (sharp[o] / (ss + 1e-8f));

  out[(size_t)b * HWSZ + gi * WW + gj] = g * accum;
}

extern "C" void kernel_launch(void* const* d_in, const int* in_sizes, int n_in,
                              void* d_out, int out_size, void* d_ws, size_t ws_size,
                              hipStream_t stream) {
  (void)in_sizes; (void)n_in; (void)out_size; (void)ws_size;
  const float* x    = (const float*)d_in[0];
  const float* gain = (const float*)d_in[1];
  const float* lrt  = (const float*)d_in[2];
  const float* rhol = (const float*)d_in[3];
  const float* Wb   = (const float*)d_in[4];
  const float* cen  = (const float*)d_in[5];
  float* out = (float*)d_out;

  float* Fm = (float*)d_ws;                    // 8 MB
  float* G  = Fm + (size_t)BATCH * HWSZ;       // 8 MB

  int total = BATCH * HWSZ;
  fm_g_kernel<<<(total + 255) / 256, 256, 0, stream>>>(x, Fm, G);

  dim3 grid(WW / 16, HH / 16, BATCH);
  dim3 blk(16, 16);
  fused_kernel<<<grid, blk, 0, stream>>>(Fm, G, Wb, cen, gain, lrt, rhol, out);
}
